// EnhancedTransformerBlock_81784767250592
// MI455X (gfx1250) — matrix operand order repaired and checked by reading, not compiled, not run on hardware
//
#include <hip/hip_runtime.h>
#include <hip/hip_bf16.h>
#include <math.h>

#define N_NODES 16384
#define BATCH   64
#define HDIM    256
#define NHEADS  8
#define HD      32
#define MAXN    512
#define EPS_GN  1e-5f

typedef _Float16 v16h __attribute__((ext_vector_type(16)));
typedef _Float16 v8h  __attribute__((ext_vector_type(8)));
typedef float    v8f  __attribute__((ext_vector_type(8)));

// ---------------------------------------------------------------------------
// WMMA helpers (A/B use contiguous k = 16*(lane/16) + s, wave32)
// ---------------------------------------------------------------------------
__device__ __forceinline__ v8f wmma_f16(v16h a, v16h b, v8f c) {
  return __builtin_amdgcn_wmma_f32_16x16x32_f16(false, a, false, b, (short)0, c,
                                                false, false);
}

// A fragment: 16x32 f16, row-major source with leading dim lda.
__device__ __forceinline__ v16h load_a_frag(const _Float16* __restrict__ A,
                                            int lda, int row0, int k0) {
  int lane = threadIdx.x & 31;
  int m    = row0 + (lane & 15);
  int hi   = lane >> 4;
  const _Float16* p = A + (size_t)m * lda + k0;
  v8h c0 = *(const v8h*)(p + (hi ? 16 : 0));
  v8h c1 = *(const v8h*)(p + (hi ? 24 : 8));
  v16h r;
#pragma unroll
  for (int i = 0; i < 8; ++i) { r[i] = c0[i]; r[8 + i] = c1[i]; }
  return r;
}

// B fragment: 32x16 f16, operand stored row-major as W[N,K].
__device__ __forceinline__ v16h load_b_frag_nk(const _Float16* __restrict__ W,
                                               int ldw, int col0, int k0) {
  int lane = threadIdx.x & 31;
  int n    = col0 + (lane & 15);
  int hi   = lane >> 4;
  const _Float16* p = W + (size_t)n * ldw + k0 + (hi ? 16 : 0);
  v8h c0 = *(const v8h*)p;
  v8h c1 = *(const v8h*)(p + 8);
  v16h r;
#pragma unroll
  for (int i = 0; i < 8; ++i) { r[i] = c0[i]; r[8 + i] = c1[i]; }
  return r;
}

// B fragment from a true row-major [K,N] matrix (P·V path).
__device__ __forceinline__ v16h load_b_frag_kn(const _Float16* __restrict__ Bm,
                                               int ldb, int col0, int k0) {
  int lane = threadIdx.x & 31;
  int n    = col0 + (lane & 15);
  int kb   = k0 + ((lane >> 4) ? 16 : 0);
  v16h r;
#pragma unroll
  for (int e = 0; e < 16; ++e) r[e] = Bm[(size_t)(kb + e) * ldb + n];
  return r;
}

// Async global -> LDS copy of 16 bytes per lane (ASYNCcnt-tracked).
__device__ __forceinline__ void async_copy_b128(const _Float16* g, _Float16* lds) {
  unsigned l = (unsigned)(uintptr_t)lds;               // LDS byte offset (AS3 ptr)
  unsigned long long ga = (unsigned long long)(uintptr_t)g;
  asm volatile("global_load_async_to_lds_b128 %0, %1, off"
               :: "v"(l), "v"(ga) : "memory");
}
__device__ __forceinline__ void wait_async(int n) {
  if (n == 0) asm volatile("s_wait_asynccnt 0x0" ::: "memory");
  else        asm volatile("s_wait_asynccnt 0x3" ::: "memory");
}

// ---------------------------------------------------------------------------
// Small setup kernels
// ---------------------------------------------------------------------------
__global__ void zero_i32(int* p, int n) {
  int i = blockIdx.x * blockDim.x + threadIdx.x;
  if (i < n) p[i] = 0;
}

__global__ void hist_kernel(const int* __restrict__ batch, int* __restrict__ counts, int n) {
  int i = blockIdx.x * blockDim.x + threadIdx.x;
  if (i < n) atomicAdd(&counts[batch[i]], 1);
}

__global__ void scan_kernel(const int* __restrict__ counts, int* __restrict__ starts) {
  if (threadIdx.x == 0) {
    int acc = 0;
    for (int b = 0; b < BATCH; ++b) { starts[b] = acc; acc += counts[b]; }
  }
}

__global__ void f32_to_f16(const float* __restrict__ s, _Float16* __restrict__ d, int n) {
  int i = blockIdx.x * blockDim.x + threadIdx.x;
  if (i < n) d[i] = (_Float16)s[i];
}

// ---------------------------------------------------------------------------
// GraphNorm fused with pack (PAD=true: padded [B,MAXN,H]; else ragged [N,H])
// ---------------------------------------------------------------------------
template <bool PAD>
__global__ __launch_bounds__(HDIM) void graphnorm_kernel(
    const float* __restrict__ x, const int* __restrict__ counts,
    const int* __restrict__ starts, const float* __restrict__ gw,
    const float* __restrict__ gb, _Float16* __restrict__ out) {
  int b = blockIdx.x;
  int h = threadIdx.x;
  int cnt = counts[b];
  int st  = starts[b];

  float s = 0.f, ss = 0.f;
  for (int j = 0; j < cnt; ++j) {
    float v = x[(size_t)(st + j) * HDIM + h];
    s += v; ss += v * v;
  }
  float cf   = (float)(cnt > 1 ? cnt : 1);
  float mean = s / cf;
  float var  = fmaxf((ss - cf * mean * mean) / fmaxf(cf - 1.f, 1.f), 0.f);
  float inv  = 1.f / (sqrtf(var) + EPS_GN);
  float wh = gw[h], bh = gb[h];

  for (int j = 0; j < cnt; ++j) {
    float v = x[(size_t)(st + j) * HDIM + h];
    float r = wh * (v - mean) * inv + bh;
    size_t o = PAD ? ((size_t)b * MAXN + j) * HDIM + h
                   : (size_t)(st + j) * HDIM + h;
    out[o] = (_Float16)r;
  }
  if (PAD) {
    for (int j = cnt; j < MAXN; ++j)
      out[((size_t)b * MAXN + j) * HDIM + h] = (_Float16)0.f;
  }
}

// ---------------------------------------------------------------------------
// WMMA GEMM, double-buffered async LDS staging:
//   Y[M,Nout] = A[M,K](f16) · W[Nout,K]^T(f16) + bias
// Block = 256 thr = 8 waves; block tile 128x64; wave tile 32x32; K-step 32.
// Tile t+1 is DMA'd (global_load_async_to_lds_b128) into the alternate LDS
// buffer while tile t feeds the WMMAs; s_wait_asynccnt 3 (in-order
// completion, 3 copies/thread/tile) gates consumption of tile t.
// EPI 0: f16 store | 1: exact GELU f16 | 2: +residual(f32), f32 store
// ---------------------------------------------------------------------------
template <int EPI>
__global__ __launch_bounds__(256) void gemm_wmma(
    const _Float16* __restrict__ A, int lda,
    const _Float16* __restrict__ W, int ldw,
    const float* __restrict__ bias, const float* __restrict__ res,
    void* __restrict__ out, int ldo, int K) {
  __shared__ __align__(16) _Float16 As[2][128 * 32];
  __shared__ __align__(16) _Float16 Bs[2][64 * 32];

  int tid  = threadIdx.x;
  int wave = tid >> 5;
  int lane = tid & 31;
  int gm0  = blockIdx.y * 128;
  int gn0  = blockIdx.x * 64;
  int wm0  = (wave >> 1) * 32;   // wave tile row within block
  int wn0  = (wave & 1) * 32;    // wave tile col within block

  // Staging decomposition: A tile = 512 x 16B chunks (2/thread), B = 256 (1/thread).
  int ca0 = tid * 2, ca1 = tid * 2 + 1;
  int ar0 = ca0 >> 2, ac0 = (ca0 & 3) * 8;
  int ar1 = ca1 >> 2, ac1 = (ca1 & 3) * 8;
  int br  = tid >> 2,  bc  = (tid & 3) * 8;

  auto stage = [&](int k0, int buf) {
    async_copy_b128(A + (size_t)(gm0 + ar0) * lda + k0 + ac0, As[buf] + ar0 * 32 + ac0);
    async_copy_b128(A + (size_t)(gm0 + ar1) * lda + k0 + ac1, As[buf] + ar1 * 32 + ac1);
    async_copy_b128(W + (size_t)(gn0 + br) * ldw + k0 + bc,  Bs[buf] + br * 32 + bc);
  };

  int nt = K / 32;
  stage(0, 0);  // prologue: tile 0 in flight

  v8f acc[2][2] = {};
  for (int t = 0; t < nt; ++t) {
    int buf = t & 1;
    if (t + 1 < nt) {
      stage((t + 1) * 32, buf ^ 1);   // overlap next tile's DMA with compute
      if (t + 2 < nt)                 // warm L2 two steps ahead
        __builtin_prefetch(A + (size_t)(gm0 + ar0) * lda + (t + 2) * 32 + ac0, 0, 1);
      wait_async(3);                  // tile t's 3 copies have landed (in-order)
    } else {
      wait_async(0);
    }
    __syncthreads();                  // tile t visible block-wide

    v16h a0 = load_a_frag(As[buf], 32, wm0, 0);
    v16h a1 = load_a_frag(As[buf], 32, wm0 + 16, 0);
    v16h b0 = load_b_frag_nk(Bs[buf], 32, wn0, 0);
    v16h b1 = load_b_frag_nk(Bs[buf], 32, wn0 + 16, 0);
    acc[0][0] = wmma_f16(a0, b0, acc[0][0]);
    acc[0][1] = wmma_f16(a0, b1, acc[0][1]);
    acc[1][0] = wmma_f16(a1, b0, acc[1][0]);
    acc[1][1] = wmma_f16(a1, b1, acc[1][1]);

    __syncthreads();                  // buf may be overwritten at t+2's stage
  }

  int hi = lane >> 4;
#pragma unroll
  for (int mi = 0; mi < 2; ++mi) {
#pragma unroll
    for (int ni = 0; ni < 2; ++ni) {
      int n = gn0 + wn0 + ni * 16 + (lane & 15);
      float bn = bias[n];
#pragma unroll
      for (int r = 0; r < 8; ++r) {
        int m = gm0 + wm0 + mi * 16 + r + hi * 8;
        float v = acc[mi][ni][r] + bn;
        size_t o = (size_t)m * ldo + n;
        if (EPI == 0) {
          ((_Float16*)out)[o] = (_Float16)v;
        } else if (EPI == 1) {
          float g = 0.5f * v * (1.f + erff(v * 0.70710678118654752f));
          ((_Float16*)out)[o] = (_Float16)g;
        } else {
          ((float*)out)[o] = res[o] + v;
        }
      }
    }
  }
}

// ---------------------------------------------------------------------------
// Attention: per (q-tile, head, graph); one wave per block.
// qkv padded f16 [B*MAXN, 3H]; ctx ragged f16 [N, H].
// ---------------------------------------------------------------------------
__global__ __launch_bounds__(32) void attention_kernel(
    const _Float16* __restrict__ qkv, const int* __restrict__ counts,
    const int* __restrict__ starts, _Float16* __restrict__ ctx) {
  __shared__ __align__(16) float    S[16 * MAXN];
  __shared__ __align__(16) _Float16 P[16 * MAXN];

  int qt = blockIdx.x, h = blockIdx.y, b = blockIdx.z;
  int cnt = counts[b];
  if (qt * 16 >= cnt) return;  // wave-uniform: EXEC all-1 at every WMMA
  int st   = starts[b];
  int lane = threadIdx.x;
  int hi   = lane >> 4;
  const int ld = 3 * HDIM;

  const _Float16* Qbase = qkv + ((size_t)b * MAXN + qt * 16) * ld + h * HD;
  v16h qf = load_a_frag(Qbase, ld, 0, 0);

  const float scale = 0.17677669529663687f;  // 1/sqrt(32)
  int jt_cnt = (cnt + 15) >> 4;
  for (int jt = 0; jt < jt_cnt; ++jt) {
    const _Float16* Kbase = qkv + ((size_t)b * MAXN + jt * 16) * ld + HDIM + h * HD;
    v16h kf = load_b_frag_nk(Kbase, ld, 0, 0);
    v8f s = {};
    s = wmma_f16(qf, kf, s);
    int n = jt * 16 + (lane & 15);
#pragma unroll
    for (int r = 0; r < 8; ++r) S[(r + hi * 8) * MAXN + n] = s[r] * scale;
  }
  __syncthreads();

  int kk = ((cnt + 31) >> 5) << 5;
  if (lane < 16) {
    int q = qt * 16 + lane;
    float*    Sr = S + lane * MAXN;
    _Float16* Pr = P + lane * MAXN;
    if (q < cnt) {
      float mx = -3.0e38f;
      for (int j = 0; j < cnt; ++j) mx = fmaxf(mx, Sr[j]);
      float sum = 0.f;
      for (int j = 0; j < cnt; ++j) { float e = expf(Sr[j] - mx); Sr[j] = e; sum += e; }
      float inv = 1.f / sum;
      for (int j = 0; j < cnt; ++j) Pr[j] = (_Float16)(Sr[j] * inv);
      for (int j = cnt; j < kk; ++j) Pr[j] = (_Float16)0.f;
    } else {
      for (int j = 0; j < kk; ++j) Pr[j] = (_Float16)0.f;
    }
  }
  __syncthreads();

  v8f acc0 = {}, acc1 = {};
  const _Float16* Vbase = qkv + (size_t)b * MAXN * ld + 2 * HDIM + h * HD;
  for (int k0 = 0; k0 < kk; k0 += 32) {
    v16h pf = load_a_frag((const _Float16*)P, MAXN, 0, k0);
    v16h v0 = load_b_frag_kn(Vbase, ld, 0, k0);
    v16h v1 = load_b_frag_kn(Vbase, ld, 16, k0);
    acc0 = wmma_f16(pf, v0, acc0);
    acc1 = wmma_f16(pf, v1, acc1);
  }

  int ncol = lane & 15;
#pragma unroll
  for (int r = 0; r < 8; ++r) {
    int q = qt * 16 + r + hi * 8;
    if (q < cnt) {
      size_t o = (size_t)(st + q) * HDIM + h * HD;
      ctx[o + ncol]      = (_Float16)acc0[r];
      ctx[o + 16 + ncol] = (_Float16)acc1[r];
    }
  }
}

// ---------------------------------------------------------------------------
// Launch orchestration
// ---------------------------------------------------------------------------
extern "C" void kernel_launch(void* const* d_in, const int* in_sizes, int n_in,
                              void* d_out, int out_size, void* d_ws, size_t ws_size,
                              hipStream_t stream) {
  const float* x    = (const float*)d_in[0];
  const int*   batch= (const int*)d_in[1];
  const float* n1w  = (const float*)d_in[3];
  const float* n1b  = (const float*)d_in[4];
  const float* win  = (const float*)d_in[5];   // [768,256]
  const float* bin  = (const float*)d_in[6];
  const float* wout = (const float*)d_in[7];   // [256,256]
  const float* bout = (const float*)d_in[8];
  const float* n2w  = (const float*)d_in[9];
  const float* n2b  = (const float*)d_in[10];
  const float* w1   = (const float*)d_in[11];  // [1024,256]
  const float* b1   = (const float*)d_in[12];
  const float* w2   = (const float*)d_in[13];  // [256,1024]
  const float* b2   = (const float*)d_in[14];
  float* out = (float*)d_out;

  char* ws = (char*)d_ws;
  size_t off = 0;
  auto alloc = [&](size_t bytes) -> void* {
    off = (off + 255) & ~(size_t)255;
    void* p = ws + off;
    off += bytes;
    return p;
  };

  int* counts        = (int*)alloc(BATCH * 4);
  int* starts        = (int*)alloc(BATCH * 4);
  _Float16* win_h    = (_Float16*)alloc((size_t)768 * 256 * 2);
  _Float16* wout_h   = (_Float16*)alloc((size_t)256 * 256 * 2);
  _Float16* w1_h     = (_Float16*)alloc((size_t)1024 * 256 * 2);
  _Float16* w2_h     = (_Float16*)alloc((size_t)256 * 1024 * 2);
  _Float16* xpad     = (_Float16*)alloc((size_t)BATCH * MAXN * HDIM * 2);
  _Float16* qkv      = (_Float16*)alloc((size_t)BATCH * MAXN * 3 * HDIM * 2);
  _Float16* ctx      = (_Float16*)alloc((size_t)N_NODES * HDIM * 2);
  float*    x2       = (float*)alloc((size_t)N_NODES * HDIM * 4);
  _Float16* xn2      = (_Float16*)alloc((size_t)N_NODES * HDIM * 2);
  _Float16* hbuf     = (_Float16*)alloc((size_t)N_NODES * 4 * HDIM * 2);

  zero_i32<<<1, 64, 0, stream>>>(counts, BATCH);
  hist_kernel<<<N_NODES / 256, 256, 0, stream>>>(batch, counts, N_NODES);
  scan_kernel<<<1, 32, 0, stream>>>(counts, starts);

  f32_to_f16<<<(768 * 256 + 255) / 256, 256, 0, stream>>>(win, win_h, 768 * 256);
  f32_to_f16<<<(256 * 256 + 255) / 256, 256, 0, stream>>>(wout, wout_h, 256 * 256);
  f32_to_f16<<<(1024 * 256 + 255) / 256, 256, 0, stream>>>(w1, w1_h, 1024 * 256);
  f32_to_f16<<<(256 * 1024 + 255) / 256, 256, 0, stream>>>(w2, w2_h, 256 * 1024);

  graphnorm_kernel<true><<<BATCH, HDIM, 0, stream>>>(x, counts, starts, n1w, n1b, xpad);

  gemm_wmma<0><<<dim3(768 / 64, (BATCH * MAXN) / 128), 256, 0, stream>>>(
      xpad, HDIM, win_h, HDIM, bin, nullptr, qkv, 3 * HDIM, HDIM);

  attention_kernel<<<dim3(MAXN / 16, NHEADS, BATCH), 32, 0, stream>>>(qkv, counts, starts, ctx);

  gemm_wmma<2><<<dim3(HDIM / 64, N_NODES / 128), 256, 0, stream>>>(
      ctx, HDIM, wout_h, HDIM, bout, x, x2, HDIM, HDIM);

  graphnorm_kernel<false><<<BATCH, HDIM, 0, stream>>>(x2, counts, starts, n2w, n2b, xn2);

  gemm_wmma<1><<<dim3(1024 / 64, N_NODES / 128), 256, 0, stream>>>(
      xn2, HDIM, w1_h, HDIM, b1, nullptr, hbuf, 4 * HDIM, HDIM);

  gemm_wmma<2><<<dim3(HDIM / 64, N_NODES / 128), 256, 0, stream>>>(
      hbuf, 4 * HDIM, w2_h, 4 * HDIM, b2, x2, out, HDIM, 4 * HDIM);
}
